// GCNModel_67851893342527
// MI455X (gfx1250) — matrix operand order repaired and checked by reading, not compiled, not run on hardware
//
#include <hip/hip_runtime.h>

// ---------------------------------------------------------------------------
// Pixel2Mesh-style GCN forward for MI455X (gfx1250, wave32, WMMA).
// Dense transforms run as bf16 WMMA GEMMs (v_wmma_f32_16x16x32_bf16) with
// f32 accumulation. Global->LDS tile staging is done by the Tensor Data
// Mover (tensor_load_to_lds, TENSORcnt) with hardware LDS row padding.
// Graph aggregation uses a deterministic CSR gather.
// ---------------------------------------------------------------------------

typedef __attribute__((ext_vector_type(16))) __bf16 v16bf;
typedef __attribute__((ext_vector_type(8)))  __bf16 v8bf;
typedef __attribute__((ext_vector_type(8)))  float  v8f;
typedef __attribute__((ext_vector_type(4)))  unsigned int v4u;
typedef __attribute__((ext_vector_type(8)))  int v8i;
typedef __attribute__((ext_vector_type(4)))  int v4i;

constexpr int Bb   = 32;
constexpr int Vv   = 2048;
constexpr int Ee   = 12288;
constexpr int IMGF = 512;
constexpr int BV   = Bb * Vv;   // 65536 rows for the GCN GEMMs
constexpr int K1P  = 544;       // 515 padded up to a multiple of 32

__device__ __forceinline__ unsigned short f32_bf16(float f) {
  unsigned int u = __float_as_uint(f);
  u += 0x7fffu + ((u >> 16) & 1u);   // round-to-nearest-even
  return (unsigned short)(u >> 16);
}

// Low 32 bits of a flat pointer to __shared__ = wave-relative LDS byte address.
__device__ __forceinline__ unsigned lds_addr(const void* p) {
  return (unsigned)(unsigned long long)p;
}

// ---------------------------------------------------------------------------
// TDM: DMA a 2-D tile [tile_d1 rows x tile_d0 elems] of 2-byte elements from
// global (row stride = row_stride_elems) into LDS, inserting 16 B of padding
// after every 64 B stored (rows are 64 B, so LDS row stride becomes 80 B).
// Rows beyond tensor_d1 are zero-filled. D# layout per CDNA5 ISA §8.3/8.4.
// This toolchain exposes the 6-arg builtin:
//   (g0 v4u, g1 v8i, g2 v4i, g3 v4i, v8i, i32 cpol)
// ---------------------------------------------------------------------------
__device__ __forceinline__ void tdm_load_tile_2d(
    unsigned lds, const void* gptr,
    unsigned tensor_d0, unsigned tensor_d1,
    unsigned tile_d0, unsigned tile_d1,
    unsigned row_stride_elems)
{
  unsigned long long ga = (unsigned long long)gptr;
  v4u g0;
  g0[0] = 1u;                                    // count=1 valid user descriptor
  g0[1] = lds;                                   // lds_addr (bytes)
  g0[2] = (unsigned)ga;                          // global_addr[31:0]
  g0[3] = (unsigned)((ga >> 32) & 0x01ffffffu)   // global_addr[56:32]
        | (2u << 30);                            // type = 2 ("image")
  v8i g1;
  g1[0] = (int)((1u << 16)                       // data_size = 2 bytes
              | (1u << 20)                       // pad_enable
              | (3u << 22)                       // pad_interval: every 16 DWORDs
              | (3u << 25));                     // pad_amount: 4 DWORDs (16 B)
  g1[1] = (int)((tensor_d0 & 0xffffu) << 16);                       // dim0[15:0]
  g1[2] = (int)((tensor_d0 >> 16) | ((tensor_d1 & 0xffffu) << 16)); // dim0[31:16]|dim1[15:0]
  g1[3] = (int)((tensor_d1 >> 16) | (tile_d0 << 16));               // dim1[31:16]|tile0
  g1[4] = (int)(tile_d1 & 0xffffu);              // tile_dim1 (tile_dim2 = 0)
  g1[5] = (int)row_stride_elems;                 // tensor_dim0_stride[31:0]
  g1[6] = 0;                                     // stride[47:32], dim1_stride lo
  g1[7] = 0;
  v4i z4; z4[0] = 0; z4[1] = 0; z4[2] = 0; z4[3] = 0;   // groups 2/3 unused (2-D)
  v8i z8;
  for (int i = 0; i < 8; ++i) z8[i] = 0;
  __builtin_amdgcn_tensor_load_to_lds(g0, g1, z4, z4, z8, 0);
}

// ---------------------------------------------------------------------------
// bf16 WMMA GEMM:  C[M,N] = A[M,Kp] * Bt[N,Kp]^T   (Bt is B pre-transposed)
// Block = 128x128x32, 256 threads = 8 waves in a 2x4 grid; each wave owns a
// 64x32 patch = 4x2 tiles of 16x16 -> 8 x v8f accumulators.
// Staging: wave 0 issues two TDM descriptors per K-step (A tile, B tile),
// double buffered; completion via s_wait_tensorcnt + workgroup barrier.
// ---------------------------------------------------------------------------
__global__ __launch_bounds__(256, 2) void wmma_gemm_bf16(
    const unsigned short* __restrict__ A,
    const unsigned short* __restrict__ Bt,
    float* __restrict__ C,
    int M, int N, int Kp)
{
  constexpr int BM = 128, BN = 128, BK = 32;
  constexpr int LDT = BK + 8;                 // LDS row stride (TDM pad engine)
  __shared__ unsigned short sA[2][BM * LDT];
  __shared__ unsigned short sB[2][BN * LDT];

  const int tid  = threadIdx.x;
  const int lane = tid & 31;
  const int wave = tid >> 5;
  const int wm   = wave & 1;                  // wave row (0..1)
  const int wn   = wave >> 1;                 // wave col (0..3)
  const int bm   = blockIdx.x * BM;
  const int bn   = blockIdx.y * BN;

  const unsigned ldsA0 = lds_addr(&sA[0][0]);
  const unsigned ldsB0 = lds_addr(&sB[0][0]);
  constexpr unsigned BUFBYTES = (unsigned)(BM * LDT * 2);

  v8f acc[4][2];
  for (int mt = 0; mt < 4; ++mt)
    for (int nt = 0; nt < 2; ++nt)
      for (int i = 0; i < 8; ++i) acc[mt][nt][i] = 0.0f;

  auto stage = [&](int buf, int k0) {
    if (wave == 0) {
      tdm_load_tile_2d(ldsA0 + (unsigned)buf * BUFBYTES,
                       A + (size_t)bm * Kp + k0,
                       (unsigned)Kp, (unsigned)(M - bm), BK, BM, (unsigned)Kp);
      tdm_load_tile_2d(ldsB0 + (unsigned)buf * BUFBYTES,
                       Bt + (size_t)bn * Kp + k0,
                       (unsigned)Kp, (unsigned)(N - bn), BK, BN, (unsigned)Kp);
    }
  };

  stage(0, 0);
  if (wave == 0) __builtin_amdgcn_s_wait_tensorcnt(0);
  __syncthreads();

  const int nk = Kp / BK;
  for (int kk = 0; kk < nk; ++kk) {
    const int cur = kk & 1;
    if (kk + 1 < nk) stage(cur ^ 1, (kk + 1) * BK);

    const int mrow = lane & 15;
    const int ka   = (lane >> 4) * 8;    // A: K = {h*8..h*8+7, 16+h*8..16+h*8+7}
    const int kb   = (lane >> 4) * 8;    // B: K = {h*8..h*8+7, 16+h*8..16+h*8+7}

    v16bf af[4], bfr[2];
    for (int mt = 0; mt < 4; ++mt) {
      const unsigned short* p = &sA[cur][(wm * 64 + mt * 16 + mrow) * LDT];
      v8bf lo = *(const v8bf*)(p + ka);
      v8bf hi = *(const v8bf*)(p + 16 + ka);
      af[mt] = __builtin_shufflevector(lo, hi, 0,1,2,3,4,5,6,7,8,9,10,11,12,13,14,15);
    }
    for (int nt = 0; nt < 2; ++nt) {
      const unsigned short* p = &sB[cur][(wn * 32 + nt * 16 + mrow) * LDT];
      v8bf lo = *(const v8bf*)(p + kb);
      v8bf hi = *(const v8bf*)(p + 16 + kb);
      bfr[nt] = __builtin_shufflevector(lo, hi, 0,1,2,3,4,5,6,7,8,9,10,11,12,13,14,15);
    }
    for (int mt = 0; mt < 4; ++mt)
      for (int nt = 0; nt < 2; ++nt)
        acc[mt][nt] = __builtin_amdgcn_wmma_f32_16x16x32_bf16(
            false, af[mt], false, bfr[nt], (short)0, acc[mt][nt], false, false);

    // DMA into the other buffer must land, and all waves must be done
    // reading `cur`, before the next iteration overwrites it.
    if (wave == 0) __builtin_amdgcn_s_wait_tensorcnt(0);
    __syncthreads();
  }

  // C/D layout: VGPR r -> M = r + 8*(lane>>4); N = lane&15 within tile.
  for (int mt = 0; mt < 4; ++mt) {
    const int row0 = bm + wm * 64 + mt * 16 + (lane >> 4) * 8;
    for (int nt = 0; nt < 2; ++nt) {
      const int col = bn + wn * 32 + nt * 16 + (lane & 15);
      if (col >= N) continue;
      for (int r = 0; r < 8; ++r) {
        const int row = row0 + r;
        if (row < M) C[(size_t)row * N + col] = acc[mt][nt][r];
      }
    }
  }
}

// ---------------------------------------------------------------------------
// Graph preprocessing
// ---------------------------------------------------------------------------
__global__ void zero_i32(int* p, int n) {
  int i = blockIdx.x * blockDim.x + threadIdx.x;
  if (i < n) p[i] = 0;
}

__global__ void count_dst(const int* __restrict__ ei, int* __restrict__ deg, int E) {
  int e = blockIdx.x * blockDim.x + threadIdx.x;
  if (e < E) atomicAdd(&deg[ei[E + e]], 1);
}

__global__ void calc_dinv(const int* __restrict__ deg, float* __restrict__ dinv, int n) {
  int i = blockIdx.x * blockDim.x + threadIdx.x;
  if (i < n) dinv[i] = rsqrtf((float)deg[i] + 1.0f);   // +1 self loop
}

// Deterministic CSR build (single thread; V=2048, E=12288 — negligible work).
__global__ void build_csr(const int* __restrict__ deg, const int* __restrict__ ei,
                          const float* __restrict__ dinv,
                          int* __restrict__ row_start,
                          int* __restrict__ csr_src, float* __restrict__ csr_w) {
  if (threadIdx.x != 0 || blockIdx.x != 0) return;
  __shared__ int cur[Vv];
  int acc = 0;
  for (int v = 0; v < Vv; ++v) { row_start[v] = acc; cur[v] = acc; acc += deg[v]; }
  row_start[Vv] = acc;
  for (int e = 0; e < Ee; ++e) {
    int s = ei[e], d = ei[Ee + e];
    int p = cur[d]++;
    csr_src[p] = s;
    csr_w[p]   = dinv[s] * dinv[d];
  }
}

// agg[b,v,f] = dinv[v]^2 * h[b,v,f] + sum_j w_j * h[b,src_j,f] + bias[f]  (opt. relu)
__global__ void gcn_aggregate(const float* __restrict__ h, float* __restrict__ xo,
                              const float* __restrict__ dinv,
                              const int* __restrict__ row_start,
                              const int* __restrict__ csr_src,
                              const float* __restrict__ csr_w,
                              const float* __restrict__ bias, int F, int relu) {
  long long i = (long long)blockIdx.x * blockDim.x + threadIdx.x;
  long long total = (long long)BV * F;
  if (i >= total) return;
  int f = (int)(i % F);
  long long bvrow = i / F;
  int v = (int)(bvrow % Vv);
  long long bbase = bvrow - v;                 // b*V
  float di = dinv[v];
  float sum = di * di * h[bvrow * F + f];
  int e0 = row_start[v], e1 = row_start[v + 1];
  for (int j = e0; j < e1; ++j)
    sum = fmaf(csr_w[j], h[(bbase + csr_src[j]) * F + f], sum);
  float val = sum + bias[f];
  if (relu) val = fmaxf(val, 0.0f);
  xo[i] = val;
}

// ---------------------------------------------------------------------------
// Layout/conversion kernels
// ---------------------------------------------------------------------------
__global__ void build_x0(const float* __restrict__ verts, const float* __restrict__ img,
                         unsigned short* __restrict__ xb, long long total) {
  long long i = (long long)blockIdx.x * blockDim.x + threadIdx.x;
  if (i >= total) return;
  int kp = (int)(i % K1P);
  long long bv = i / K1P;
  int b = (int)(bv / Vv);
  float v = 0.0f;
  if (kp < 3)              v = verts[bv * 3 + kp];
  else if (kp < 3 + IMGF)  v = img[(long long)b * IMGF + (kp - 3)];
  xb[i] = f32_bf16(v);
}

// W[K,N] fp32 -> Wt[N,Kp] bf16 (zero pad K..Kp)
__global__ void transpose_w_bf16(const float* __restrict__ W, unsigned short* __restrict__ Wt,
                                 int K, int N, int Kp) {
  long long i = (long long)blockIdx.x * blockDim.x + threadIdx.x;
  if (i >= (long long)N * Kp) return;
  int kp = (int)(i % Kp);
  int n  = (int)(i / Kp);
  float v = (kp < K) ? W[(long long)kp * N + n] : 0.0f;
  Wt[i] = f32_bf16(v);
}

__global__ void to_bf16(const float* __restrict__ x, unsigned short* __restrict__ y, long long n) {
  long long i = (long long)blockIdx.x * blockDim.x + threadIdx.x;
  if (i < n) y[i] = f32_bf16(x[i]);
}

__global__ void bias_to_bf16(const float* __restrict__ x, const float* __restrict__ bias,
                             unsigned short* __restrict__ y, int rows, int cols) {
  long long i = (long long)blockIdx.x * blockDim.x + threadIdx.x;
  if (i >= (long long)rows * cols) return;
  y[i] = f32_bf16(x[i] + bias[i % cols]);
}

// Tiny layer 6 transform: [BV,64] x W6[64,3] -> [BV,3] (fp32 VALU; K too small for WMMA)
__global__ void dense_64x3(const float* __restrict__ x, const float* __restrict__ W,
                           float* __restrict__ h) {
  long long i = (long long)blockIdx.x * blockDim.x + threadIdx.x;
  if (i >= (long long)BV * 3) return;
  int o = (int)(i % 3);
  long long row = i / 3;
  const float* xr = x + row * 64;
  float s = 0.0f;
  for (int k = 0; k < 64; ++k) s = fmaf(xr[k], W[k * 3 + o], s);
  h[i] = s;
}

__global__ void final_out(const float* __restrict__ verts, const float* __restrict__ t3,
                          const float* __restrict__ fcb3, float* __restrict__ out) {
  long long i = (long long)blockIdx.x * blockDim.x + threadIdx.x;
  if (i >= (long long)Bb * Vv * 3) return;
  int col = (int)(i % (Vv * 3));
  out[i] = verts[i] + 0.1f * tanhf(t3[i] + fcb3[col]);
}

// ---------------------------------------------------------------------------
extern "C" void kernel_launch(void* const* d_in, const int* in_sizes, int n_in,
                              void* d_out, int out_size, void* d_ws, size_t ws_size,
                              hipStream_t stream) {
  (void)in_sizes; (void)n_in; (void)out_size; (void)ws_size;
  const float* verts = (const float*)d_in[0];
  const float* img   = (const float*)d_in[1];
  const int*   ei    = (const int*)d_in[2];
  const float* W[6];  const float* bvec[6];
  for (int i = 0; i < 6; ++i) { W[i] = (const float*)d_in[3 + 2 * i]; bvec[i] = (const float*)d_in[4 + 2 * i]; }
  const float* fcW[3]; const float* fcb[3];
  for (int i = 0; i < 3; ++i) { fcW[i] = (const float*)d_in[15 + 2 * i]; fcb[i] = (const float*)d_in[16 + 2 * i]; }
  float* out = (float*)d_out;

  // --- workspace carve-out (256B aligned) ---
  char* p = (char*)d_ws;
  auto alloc = [&](size_t bytes) -> char* {
    char* r = p; p += (bytes + 255) & ~(size_t)255; return r;
  };
  float* dinv      = (float*)alloc((size_t)Vv * 4);
  int*   deg       = (int*)  alloc((size_t)Vv * 4);
  int*   row_start = (int*)  alloc((size_t)(Vv + 1) * 4);
  int*   csr_src   = (int*)  alloc((size_t)Ee * 4);
  float* csr_w     = (float*)alloc((size_t)Ee * 4);
  unsigned short* xb = (unsigned short*)alloc((size_t)BV * K1P * 2);
  float* hbuf = (float*)alloc((size_t)BV * 512 * 4);
  float* xbuf = (float*)alloc((size_t)BV * 512 * 4);
  unsigned short* Wt[5];
  Wt[0] = (unsigned short*)alloc((size_t)512 * K1P * 2);
  Wt[1] = (unsigned short*)alloc((size_t)512 * 512 * 2);
  Wt[2] = (unsigned short*)alloc((size_t)512 * 512 * 2);
  Wt[3] = (unsigned short*)alloc((size_t)512 * 512 * 2);
  Wt[4] = (unsigned short*)alloc((size_t)64  * 512 * 2);
  unsigned short* fcWt[3];
  fcWt[0] = (unsigned short*)alloc((size_t)1024 * 6144 * 2);
  fcWt[1] = (unsigned short*)alloc((size_t)1024 * 1024 * 2);
  fcWt[2] = (unsigned short*)alloc((size_t)6144 * 1024 * 2);
  float*          tfp = (float*)         alloc((size_t)Bb * 6144 * 4);
  unsigned short* tb  = (unsigned short*)alloc((size_t)Bb * 6144 * 2);

  const int T = 256;
  auto cdiv = [](long long a, long long b) { return (int)((a + b - 1) / b); };
  auto gemm = [&](const unsigned short* A, const unsigned short* Bt, float* C,
                  int M, int N, int Kp) {
    dim3 g((M + 127) / 128, (N + 127) / 128);
    wmma_gemm_bf16<<<g, 256, 0, stream>>>(A, Bt, C, M, N, Kp);
  };
  auto agg = [&](const float* h, float* xo, const float* bias, int F, int relu) {
    gcn_aggregate<<<cdiv((long long)BV * F, T), T, 0, stream>>>(
        h, xo, dinv, row_start, csr_src, csr_w, bias, F, relu);
  };

  // graph prep
  zero_i32 <<<cdiv(Vv, T), T, 0, stream>>>(deg, Vv);
  count_dst<<<cdiv(Ee, T), T, 0, stream>>>(ei, deg, Ee);
  calc_dinv<<<cdiv(Vv, T), T, 0, stream>>>(deg, dinv, Vv);
  build_csr<<<1, 32, 0, stream>>>(deg, ei, dinv, row_start, csr_src, csr_w);

  // bf16 transposed weights
  transpose_w_bf16<<<cdiv((long long)512 * K1P, T), T, 0, stream>>>(W[0], Wt[0], 515, 512, K1P);
  transpose_w_bf16<<<cdiv((long long)512 * 512, T), T, 0, stream>>>(W[1], Wt[1], 512, 512, 512);
  transpose_w_bf16<<<cdiv((long long)512 * 512, T), T, 0, stream>>>(W[2], Wt[2], 512, 512, 512);
  transpose_w_bf16<<<cdiv((long long)512 * 512, T), T, 0, stream>>>(W[3], Wt[3], 512, 512, 512);
  transpose_w_bf16<<<cdiv((long long)64  * 512, T), T, 0, stream>>>(W[4], Wt[4], 512, 64, 512);
  transpose_w_bf16<<<cdiv((long long)1024 * 6144, T), T, 0, stream>>>(fcW[0], fcWt[0], 6144, 1024, 6144);
  transpose_w_bf16<<<cdiv((long long)1024 * 1024, T), T, 0, stream>>>(fcW[1], fcWt[1], 1024, 1024, 1024);
  transpose_w_bf16<<<cdiv((long long)6144 * 1024, T), T, 0, stream>>>(fcW[2], fcWt[2], 1024, 6144, 1024);

  // x0 = [verts | img] padded to 544, bf16
  build_x0<<<cdiv((long long)BV * K1P, T), T, 0, stream>>>(verts, img, xb, (long long)BV * K1P);

  // GCN layers 1..4 (512-wide)
  gemm(xb, Wt[0], hbuf, BV, 512, K1P);
  agg(hbuf, xbuf, bvec[0], 512, 0);
  to_bf16<<<cdiv((long long)BV * 512, T), T, 0, stream>>>(xbuf, xb, (long long)BV * 512);

  gemm(xb, Wt[1], hbuf, BV, 512, 512);
  agg(hbuf, xbuf, bvec[1], 512, 1);
  to_bf16<<<cdiv((long long)BV * 512, T), T, 0, stream>>>(xbuf, xb, (long long)BV * 512);

  gemm(xb, Wt[2], hbuf, BV, 512, 512);
  agg(hbuf, xbuf, bvec[2], 512, 0);
  to_bf16<<<cdiv((long long)BV * 512, T), T, 0, stream>>>(xbuf, xb, (long long)BV * 512);

  gemm(xb, Wt[3], hbuf, BV, 512, 512);
  agg(hbuf, xbuf, bvec[3], 512, 1);
  to_bf16<<<cdiv((long long)BV * 512, T), T, 0, stream>>>(xbuf, xb, (long long)BV * 512);

  // layer 5: 512 -> 64
  gemm(xb, Wt[4], hbuf, BV, 64, 512);
  agg(hbuf, xbuf, bvec[4], 64, 0);

  // layer 6: 64 -> 3 (scalar), aggregate + relu
  dense_64x3<<<cdiv((long long)BV * 3, T), T, 0, stream>>>(xbuf, W[5], hbuf);
  agg(hbuf, xbuf, bvec[5], 3, 1);

  // FC head: x6 [32, 6144]
  to_bf16<<<cdiv((long long)Bb * 6144, T), T, 0, stream>>>(xbuf, tb, (long long)Bb * 6144);
  gemm(tb, fcWt[0], tfp, Bb, 1024, 6144);
  bias_to_bf16<<<cdiv((long long)Bb * 1024, T), T, 0, stream>>>(tfp, fcb[0], tb, Bb, 1024);
  gemm(tb, fcWt[1], tfp, Bb, 1024, 1024);
  bias_to_bf16<<<cdiv((long long)Bb * 1024, T), T, 0, stream>>>(tfp, fcb[1], tb, Bb, 1024);
  gemm(tb, fcWt[2], tfp, Bb, 6144, 1024);

  final_out<<<cdiv((long long)Bb * 6144, T), T, 0, stream>>>(verts, tfp, fcb[2], out);
}
